// _ClapSwinBlock_3427383903075
// MI455X (gfx1250) — compile-verified, matrix operand order checked by reading, not run on hardware
//
#include <hip/hip_runtime.h>
#include <hip/hip_bf16.h>

typedef __attribute__((ext_vector_type(16))) _Float16 v16h;
typedef __attribute__((ext_vector_type(8)))  _Float16 v8h;
typedef __attribute__((ext_vector_type(8)))  float    v8f;

#define CDIM 256
#define DFF  1024
#define BATCH 32
#define HDIM 64
#define WDIM 64
#define WS 8
#define SS 4
#define NH 8
#define NWIN 64          // (64/8)*(64/8)
#define NTOK 64          // tokens per window
#define TOKENS (BATCH * HDIM * WDIM)   // 131072
#define NBM (NWIN * NH * 16 * 32 * 8)  // combined bias+mask table elements (2M)

// ---------------------------------------------------------------------------
// Fragment loaders matching the CDNA5 16-bit WMMA VGPR layouts (ISA 7.12.2):
// element i of v16h on lane L maps to K = (i&7) + (i>>3)*16 + (L>=16 ? 8 : 0),
// row/col = L&15.  Two contiguous 8-half (16B) loads per lane.
// ---------------------------------------------------------------------------
__device__ __forceinline__ v16h frag_at(const _Float16* base, int k0) {
  union { v16h v; v8h h[2]; } u;
  u.h[0] = *(const v8h*)(base + k0);
  u.h[1] = *(const v8h*)(base + k0 + 16);
  return u.v;
}

__device__ __forceinline__ v16h load_frag(const _Float16* p, int row0, int col0,
                                          int ld, int lane) {
  int m  = lane & 15;
  int hi = lane >> 4;
  return frag_at(p + (size_t)(row0 + m) * ld + hi * 8, col0);
}

// B fragment where the K dimension is strided in memory (B[k][n] with row
// stride ld): 16 scalar 2-byte loads per lane.
__device__ __forceinline__ v16h load_fragB_strided(const _Float16* p, int krow0,
                                                   int col0, int ld, int lane) {
  int n  = lane & 15;
  int hi = lane >> 4;
  v16h out;
#pragma unroll
  for (int i = 0; i < 16; ++i) {
    int k = (i & 7) + ((i >> 3) * 16) + hi * 8;
    out[i] = p[(size_t)(krow0 + k) * ld + col0 + n];
  }
  return out;
}

__device__ __forceinline__ float rsum32(float x) {
#pragma unroll
  for (int m = 1; m <= 16; m <<= 1) x += __shfl_xor(x, m, 32);
  return x;
}
__device__ __forceinline__ float rsum16(float x) {
#pragma unroll
  for (int m = 1; m <= 8; m <<= 1) x += __shfl_xor(x, m, 32);
  return x;
}
__device__ __forceinline__ float rmax16(float x) {
#pragma unroll
  for (int m = 1; m <= 8; m <<= 1) x = fmaxf(x, __shfl_xor(x, m, 32));
  return x;
}

// region label for the Swin shift mask (H-WS=56, H-SS=60)
__device__ __forceinline__ int region1d(int c) {
  return (c < HDIM - WS) ? 0 : ((c < HDIM - SS) ? 1 : 2);
}

// ---------------------------------------------------------------------------
// Weight convert: f32 [K,N] -> f16 transposed [N,K]
// ---------------------------------------------------------------------------
__global__ void k_wconv(const float* __restrict__ W, _Float16* __restrict__ Wt,
                        int K, int N) {
  int t = blockIdx.x * 256 + threadIdx.x;
  if (t < K * N) {
    int k = t / N, n = t % N;
    Wt[(size_t)n * K + k] = (_Float16)W[t];
  }
}

// ---------------------------------------------------------------------------
// Combined relative-position bias + shift mask table, laid out in WMMA
// C-fragment order: comb[wi][h][tile(mt*4+nt)][lane][r].  (Batch-invariant:
// only 64 windows x 8 heads x 64x64 = 8 MB, L2-resident, built once.)
// Each lane of the attention kernel then fetches its 8 addends for a 16x16
// tile as one contiguous v8f load.
// ---------------------------------------------------------------------------
__global__ void k_biasmask(const float* __restrict__ bias_table,
                           float* __restrict__ comb) {
  int t = blockIdx.x * 256 + threadIdx.x;   // NBM / 256 blocks, exact
  int r    = t & 7;
  int lane = (t >> 3) & 31;
  int tile = (t >> 8) & 15;
  int h    = (t >> 12) & 7;
  int wi   = t >> 15;
  int mt = tile >> 2, nt = tile & 3;
  int hi = lane >> 4, nn = lane & 15;
  int whi = wi >> 3, wwi = wi & 7;

  int irow = 16 * mt + r + 8 * hi;
  int jcol = 16 * nt + nn;
  int yi = irow >> 3, xi = irow & 7;
  int yj = jcol >> 3, xj = jcol & 7;
  int ridx = (yi - yj + WS - 1) * (2 * WS - 1) + (xi - xj + WS - 1);
  float bb = bias_table[ridx * NH + h];
  int ri = region1d(whi * WS + yi) * 3 + region1d(wwi * WS + xi);
  int rj = region1d(whi * WS + yj) * 3 + region1d(wwi * WS + xj);
  comb[t] = bb + ((ri != rj) ? -100.0f : 0.0f);
}

// ---------------------------------------------------------------------------
// LN1 + cyclic shift + window partition.  One wave per destination row.
// ---------------------------------------------------------------------------
__global__ void k_ln1_window(const float* __restrict__ x,
                             const float* __restrict__ g,
                             const float* __restrict__ bb,
                             _Float16* __restrict__ out) {
  int lane = threadIdx.x & 31;
  int wv   = threadIdx.x >> 5;
  int r    = blockIdx.x * 8 + wv;           // destination row (window order)
  int b    = r >> 12;                        // / (NWIN*NTOK)
  int rem  = r & 4095;
  int wi   = rem >> 6;
  int ti   = rem & 63;
  int hh   = (wi >> 3) * WS + (ti >> 3);
  int ww   = (wi & 7) * WS + (ti & 7);
  int sh   = (hh + SS) & (HDIM - 1);
  int sw   = (ww + SS) & (WDIM - 1);
  const float* src = x + (((size_t)b * HDIM + sh) * WDIM + sw) * CDIM;

  float v[8], s = 0.f, s2 = 0.f;
#pragma unroll
  for (int j = 0; j < 8; ++j) {
    v[j] = src[lane + 32 * j];
    s += v[j];
    s2 += v[j] * v[j];
  }
  s  = rsum32(s);
  s2 = rsum32(s2);
  float mean = s * (1.f / CDIM);
  float var  = s2 * (1.f / CDIM) - mean * mean;
  float rstd = rsqrtf(var + 1e-5f);
#pragma unroll
  for (int j = 0; j < 8; ++j) {
    int c = lane + 32 * j;
    out[(size_t)r * CDIM + c] = (_Float16)((v[j] - mean) * rstd * g[c] + bb[c]);
  }
}

// LN2: token order, f32 in -> f16 out
__global__ void k_ln2(const float* __restrict__ x, const float* __restrict__ g,
                      const float* __restrict__ bb, _Float16* __restrict__ out) {
  int lane = threadIdx.x & 31;
  int wv   = threadIdx.x >> 5;
  int r    = blockIdx.x * 8 + wv;
  const float* src = x + (size_t)r * CDIM;
  float v[8], s = 0.f, s2 = 0.f;
#pragma unroll
  for (int j = 0; j < 8; ++j) {
    v[j] = src[lane + 32 * j];
    s += v[j];
    s2 += v[j] * v[j];
  }
  s  = rsum32(s);
  s2 = rsum32(s2);
  float mean = s * (1.f / CDIM);
  float rstd = rsqrtf(s2 * (1.f / CDIM) - mean * mean + 1e-5f);
#pragma unroll
  for (int j = 0; j < 8; ++j) {
    int c = lane + 32 * j;
    out[(size_t)r * CDIM + c] = (_Float16)((v[j] - mean) * rstd * g[c] + bb[c]);
  }
}

// ---------------------------------------------------------------------------
// WMMA GEMM: out[TOKENS,N] = A[TOKENS,K](f16) * Bt[N,K]^T(f16) + bias, fused
// epilogues.  N, K, MODE are compile-time so the k-loop carries no address
// arithmetic: 6 per-lane base pointers are formed once and every
// global_load_b128 in the loop uses an immediate offset.  One wave owns a
// 32x64 panel (8 WMMAs per k-step from 6 fragment loads); 8 waves/block share
// the 64-col B panel; `#pragma unroll 2` lets the scheduler hoist the next
// k-step's 12 b128 loads above the current WMMAs (partial s_wait_loadcnt).
// __launch_bounds__(256,1): optimize for 1 block/CU so the allocator keeps
// all fragments + accumulators in VGPRs (no scratch spills).
// MODE 0: f16 out            MODE 1: f16 GELU out
// MODE 2: f32 hs[remap] = res[remap] + val   (attention out-proj)
// MODE 3: f32 out[row]  = res[row]  + val    (MLP2 + final residual)
// ---------------------------------------------------------------------------
template <int N, int K, int MODE>
__global__ __launch_bounds__(256, 1) void k_gemm(
    const _Float16* __restrict__ A, const _Float16* __restrict__ Bt,
    const float* __restrict__ bias, _Float16* __restrict__ outH,
    float* __restrict__ outF, const float* __restrict__ res) {
  constexpr int TILES_M = TOKENS >> 5;       // 4096 (power of two)
  int lane = threadIdx.x & 31;
  int wv   = threadIdx.x >> 5;
  int id   = blockIdx.x * 8 + wv;
  int tm   = id & (TILES_M - 1);
  int tg   = id / TILES_M;
  int m0   = tm << 5;
  int n0   = tg << 6;

  int m  = lane & 15;
  int hi = lane >> 4;

  // per-lane fragment base pointers (loop-invariant)
  const _Float16* aB[2];
  aB[0] = A + (size_t)(m0 + m) * K + hi * 8;
  aB[1] = aB[0] + (size_t)16 * K;
  const _Float16* bB[4];
#pragma unroll
  for (int j = 0; j < 4; ++j)
    bB[j] = Bt + (size_t)(n0 + 16 * j + m) * K + hi * 8;

  v8f acc[2][4];
#pragma unroll
  for (int mt = 0; mt < 2; ++mt)
#pragma unroll
    for (int j = 0; j < 4; ++j) acc[mt][j] = v8f{0};

#pragma unroll 2
  for (int k0 = 0; k0 < K; k0 += 32) {
    v16h a[2], b[4];
#pragma unroll
    for (int mt = 0; mt < 2; ++mt) a[mt] = frag_at(aB[mt], k0);
#pragma unroll
    for (int j = 0; j < 4; ++j) b[j] = frag_at(bB[j], k0);
#pragma unroll
    for (int j = 0; j < 4; ++j) {
#pragma unroll
      for (int mt = 0; mt < 2; ++mt)
        acc[mt][j] = __builtin_amdgcn_wmma_f32_16x16x32_f16(
            false, a[mt], false, b[j], (short)0, acc[mt][j], false, false);
    }
  }

  int nn = lane & 15;
#pragma unroll
  for (int mt = 0; mt < 2; ++mt) {
#pragma unroll
    for (int j = 0; j < 4; ++j) {
#pragma unroll
      for (int r = 0; r < 8; ++r) {
        int row = m0 + 16 * mt + r + 8 * hi;
        int col = n0 + 16 * j + nn;
        float val = acc[mt][j][r] + bias[col];
        if (MODE == 0) {
          outH[(size_t)row * N + col] = (_Float16)val;
        } else if (MODE == 1) {
          float gl = 0.5f * val * (1.0f + erff(val * 0.70710678118654752f));
          outH[(size_t)row * N + col] = (_Float16)gl;
        } else if (MODE == 2) {
          // window-reverse + roll(+SS): scatter to (hh+SS)%H,(ww+SS)%W
          int bb2 = row >> 12;
          int rem = row & 4095;
          int wi  = rem >> 6;
          int ti  = rem & 63;
          int hh  = (wi >> 3) * WS + (ti >> 3);
          int ww  = (wi & 7) * WS + (ti & 7);
          int dh  = (hh + SS) & (HDIM - 1);
          int dw  = (ww + SS) & (WDIM - 1);
          size_t dst = (((size_t)bb2 * HDIM + dh) * WDIM + dw) * CDIM + col;
          outF[dst] = res[dst] + val;
        } else {  // MODE 3
          size_t dst = (size_t)row * CDIM + col;
          outF[dst] = res[dst] + val;
        }
      }
    }
  }
}

// ---------------------------------------------------------------------------
// Windowed attention: block = one window (B*nW = 2048), wave = one head.
// S = QK^T + softmax processed as two 32-row halves written as straight-line
// code (fully unrolled: no loop phis -> no register rotation copies; each
// half's accumulators die at their LDS stores).  Bias+mask come from the
// precomputed fragment-ordered table (one v8f load per 16x16 tile per lane).
// P is staged in LDS in A-fragment-addressable form, then O = P V.
// ---------------------------------------------------------------------------
__global__ __launch_bounds__(256, 1) void k_attention(
    const _Float16* __restrict__ q, const _Float16* __restrict__ k,
    const _Float16* __restrict__ v, const float* __restrict__ comb,
    _Float16* __restrict__ ao) {
  __shared__ _Float16 Psh[NH * NTOK * NTOK];  // 64 KB

  int lane = threadIdx.x & 31;
  int h    = threadIdx.x >> 5;
  int win  = blockIdx.x;
  int wi   = win & (NWIN - 1);
  int hc   = h * 32;
  int hi   = lane >> 4, nn = lane & 15;

  const _Float16* Q  = q + (size_t)win * NTOK * CDIM + hc;
  const _Float16* Kp = k + (size_t)win * NTOK * CDIM + hc;
  const _Float16* Vp = v + (size_t)win * NTOK * CDIM + hc;
  // fragment-ordered bias+mask slice for (wi, h): 16 tiles x 32 lanes x 8
  const float* cb = comb + (((size_t)wi * NH + h) * 16 * 32 + lane) * 8;

  const float scale = 0.17677669529663687f;  // 32^-0.5

  // ---- S = Q K^T + softmax, two independent 32-row halves (straight-line)
#pragma unroll
  for (int half = 0; half < 2; ++half) {
    v8f s[2][4];
#pragma unroll
    for (int mt = 0; mt < 2; ++mt)
#pragma unroll
      for (int nt = 0; nt < 4; ++nt) s[mt][nt] = v8f{0};

    v16h aq[2];
#pragma unroll
    for (int mt = 0; mt < 2; ++mt)
      aq[mt] = load_frag(Q, 16 * (2 * half + mt), 0, CDIM, lane);
#pragma unroll
    for (int nt = 0; nt < 4; ++nt) {
      v16h bk = load_frag(Kp, 16 * nt, 0, CDIM, lane);
#pragma unroll
      for (int mt = 0; mt < 2; ++mt)
        s[mt][nt] = __builtin_amdgcn_wmma_f32_16x16x32_f16(
            false, aq[mt], false, bk, (short)0, s[mt][nt], false, false);
    }

    // scale + (bias+mask) from fragment-ordered table
#pragma unroll
    for (int mt = 0; mt < 2; ++mt) {
      int mtg = 2 * half + mt;
#pragma unroll
      for (int nt = 0; nt < 4; ++nt) {
        v8f bm = *(const v8f*)(cb + (size_t)(mtg * 4 + nt) * 32 * 8);
#pragma unroll
        for (int r = 0; r < 8; ++r)
          s[mt][nt][r] = s[mt][nt][r] * scale + bm[r];
      }
    }

    // row softmax (rows live across 16-lane halves + 4 col-tiles)
#pragma unroll
    for (int mt = 0; mt < 2; ++mt) {
      int mtg = 2 * half + mt;
#pragma unroll
      for (int r = 0; r < 8; ++r) {
        float mx = -1e30f;
#pragma unroll
        for (int nt = 0; nt < 4; ++nt) mx = fmaxf(mx, s[mt][nt][r]);
        mx = rmax16(mx);
        float sum = 0.f;
#pragma unroll
        for (int nt = 0; nt < 4; ++nt) {
          float e = __expf(s[mt][nt][r] - mx);
          s[mt][nt][r] = e;
          sum += e;
        }
        sum = rsum16(sum);
        float inv = 1.0f / sum;
        int irow = 16 * mtg + r + 8 * hi;
#pragma unroll
        for (int nt = 0; nt < 4; ++nt)
          Psh[((h << 6) + irow) * NTOK + 16 * nt + nn] =
              (_Float16)(s[mt][nt][r] * inv);
      }
    }
  }
  __syncthreads();

  // ---- O = P V ----
  const _Float16* Pl = &Psh[(size_t)h * NTOK * NTOK];
  v8f o[4][2];
#pragma unroll
  for (int mt = 0; mt < 4; ++mt)
#pragma unroll
    for (int ct = 0; ct < 2; ++ct) o[mt][ct] = v8f{0};

#pragma unroll
  for (int kt = 0; kt < 2; ++kt) {
    v16h bv[2];
#pragma unroll
    for (int ct = 0; ct < 2; ++ct)
      bv[ct] = load_fragB_strided(Vp, 32 * kt, 16 * ct, CDIM, lane);
#pragma unroll
    for (int mt = 0; mt < 4; ++mt) {
      v16h ap = load_frag(Pl, 16 * mt, 32 * kt, NTOK, lane);
#pragma unroll
      for (int ct = 0; ct < 2; ++ct)
        o[mt][ct] = __builtin_amdgcn_wmma_f32_16x16x32_f16(false, ap, false,
                                                           bv[ct], (short)0,
                                                           o[mt][ct], false,
                                                           false);
    }
  }

#pragma unroll
  for (int mt = 0; mt < 4; ++mt) {
#pragma unroll
    for (int ct = 0; ct < 2; ++ct) {
#pragma unroll
      for (int r = 0; r < 8; ++r) {
        int row = 16 * mt + r + 8 * hi;
        int col = hc + 16 * ct + nn;
        ao[((size_t)win * NTOK + row) * CDIM + col] = (_Float16)o[mt][ct][r];
      }
    }
  }
}

// ---------------------------------------------------------------------------
extern "C" void kernel_launch(void* const* d_in, const int* in_sizes, int n_in,
                              void* d_out, int out_size, void* d_ws,
                              size_t ws_size, hipStream_t stream) {
  const float* hidden = (const float*)d_in[0];
  const float* ln1_g = (const float*)d_in[1];
  const float* ln1_b = (const float*)d_in[2];
  const float* wq = (const float*)d_in[3];
  const float* bq = (const float*)d_in[4];
  const float* wk = (const float*)d_in[5];
  const float* bk = (const float*)d_in[6];
  const float* wv = (const float*)d_in[7];
  const float* bv = (const float*)d_in[8];
  const float* bias_table = (const float*)d_in[9];
  const float* wo = (const float*)d_in[10];
  const float* bo = (const float*)d_in[11];
  const float* ln2_g = (const float*)d_in[12];
  const float* ln2_b = (const float*)d_in[13];
  const float* w1 = (const float*)d_in[14];
  const float* b1 = (const float*)d_in[15];
  const float* w2 = (const float*)d_in[16];
  const float* b2 = (const float*)d_in[17];
  float* out = (float*)d_out;

  char* ws = (char*)d_ws;
  size_t off = 0;
  auto alloc = [&](size_t bytes) {
    size_t o = off;
    off = (off + bytes + 255) & ~(size_t)255;
    return o;
  };
  const size_t actH = (size_t)TOKENS * CDIM * sizeof(_Float16);  // 64 MB
  size_t o_wqT = alloc(CDIM * CDIM * 2);
  size_t o_wkT = alloc(CDIM * CDIM * 2);
  size_t o_wvT = alloc(CDIM * CDIM * 2);
  size_t o_woT = alloc(CDIM * CDIM * 2);
  size_t o_w1T = alloc(CDIM * DFF * 2);
  size_t o_w2T = alloc(DFF * CDIM * 2);
  size_t o_cb = alloc((size_t)NBM * sizeof(float));             // 8 MB
  size_t o_xw = alloc(actH);  // LN1 windows; later reused for LN2 output
  size_t o_q = alloc(actH);   // q,k,v,ao contiguous (256 MB) -> aliased by y1
  size_t o_k = alloc(actH);
  size_t o_v = alloc(actH);
  size_t o_ao = alloc(actH);
  size_t o_hs = alloc((size_t)TOKENS * CDIM * sizeof(float));   // 128 MB
  size_t o_y1 = o_q;  // MLP hidden f16 (256 MB) aliases retired q/k/v/ao

  _Float16* wqT = (_Float16*)(ws + o_wqT);
  _Float16* wkT = (_Float16*)(ws + o_wkT);
  _Float16* wvT = (_Float16*)(ws + o_wvT);
  _Float16* woT = (_Float16*)(ws + o_woT);
  _Float16* w1T = (_Float16*)(ws + o_w1T);
  _Float16* w2T = (_Float16*)(ws + o_w2T);
  float* cb = (float*)(ws + o_cb);
  _Float16* xw = (_Float16*)(ws + o_xw);
  _Float16* qb = (_Float16*)(ws + o_q);
  _Float16* kb = (_Float16*)(ws + o_k);
  _Float16* vb = (_Float16*)(ws + o_v);
  _Float16* aob = (_Float16*)(ws + o_ao);
  float* hs = (float*)(ws + o_hs);
  _Float16* y1 = (_Float16*)(ws + o_y1);
  _Float16* ln2h = xw;

  // 1) weight conversion (f32 [K,N] -> f16 [N,K]) + bias/mask table
  k_wconv<<<(CDIM * CDIM + 255) / 256, 256, 0, stream>>>(wq, wqT, CDIM, CDIM);
  k_wconv<<<(CDIM * CDIM + 255) / 256, 256, 0, stream>>>(wk, wkT, CDIM, CDIM);
  k_wconv<<<(CDIM * CDIM + 255) / 256, 256, 0, stream>>>(wv, wvT, CDIM, CDIM);
  k_wconv<<<(CDIM * CDIM + 255) / 256, 256, 0, stream>>>(wo, woT, CDIM, CDIM);
  k_wconv<<<(CDIM * DFF + 255) / 256, 256, 0, stream>>>(w1, w1T, CDIM, DFF);
  k_wconv<<<(DFF * CDIM + 255) / 256, 256, 0, stream>>>(w2, w2T, DFF, CDIM);
  k_biasmask<<<NBM / 256, 256, 0, stream>>>(bias_table, cb);

  // 2) LN1 + shift + window partition
  k_ln1_window<<<TOKENS / 8, 256, 0, stream>>>(hidden, ln1_g, ln1_b, xw);

  // 3) Q,K,V projections (32x64 wave panels: M/32 * N/64 tiles, 8 per block)
  const int gQKV = (TOKENS / 32) * (CDIM / 64) / 8;   // 2048 blocks
  k_gemm<CDIM, CDIM, 0><<<gQKV, 256, 0, stream>>>(xw, wqT, bq, qb, nullptr,
                                                  nullptr);
  k_gemm<CDIM, CDIM, 0><<<gQKV, 256, 0, stream>>>(xw, wkT, bk, kb, nullptr,
                                                  nullptr);
  k_gemm<CDIM, CDIM, 0><<<gQKV, 256, 0, stream>>>(xw, wvT, bv, vb, nullptr,
                                                  nullptr);

  // 4) windowed attention (2048 windows, 8 heads = 8 waves per block)
  k_attention<<<BATCH * NWIN, 256, 0, stream>>>(qb, kb, vb, cb, aob);

  // 5) output projection + window reverse + roll + residual -> hs (f32)
  k_gemm<CDIM, CDIM, 2><<<gQKV, 256, 0, stream>>>(aob, woT, bo, nullptr, hs,
                                                  hidden);

  // 6) LN2 -> f16
  k_ln2<<<TOKENS / 8, 256, 0, stream>>>(hs, ln2_g, ln2_b, ln2h);

  // 7) MLP GEMM1 + exact GELU
  const int gFF1 = (TOKENS / 32) * (DFF / 64) / 8;    // 8192 blocks
  k_gemm<DFF, CDIM, 1><<<gFF1, 256, 0, stream>>>(ln2h, w1T, b1, y1, nullptr,
                                                 nullptr);

  // 8) MLP GEMM2 + final residual -> d_out (f32)
  k_gemm<CDIM, DFF, 3><<<gQKV, 256, 0, stream>>>(y1, w2T, b2, nullptr, out,
                                                 hs);
}
